// RTMCCBlock_1563368096603
// MI455X (gfx1250) — matrix operand order repaired and checked by reading, not compiled, not run on hardware
//
#include <hip/hip_runtime.h>
#include <hip/hip_bf16.h>

typedef __attribute__((ext_vector_type(16))) _Float16 v16h;
typedef __attribute__((ext_vector_type(8)))  _Float16 v8h;
typedef __attribute__((ext_vector_type(8)))  float    v8f;
typedef int vi4 __attribute__((vector_size(16)));     // b128 payload type for async-LDS builtin

#define Dm   256
#define Em   512
#define Sm   128
#define Tm   133
#define TP   144            // padded T (9 tiles of 16)
#define KERNP 160           // kernel-matrix row pitch (padded for last K-step)
#define RSQRT_S 0.08838834764831845f

// ---- LDS region offsets (bytes) ----
// [0      , 73728 )  xn[144][256] f16   (stage1) / kernel[144][160] f16 (stage2+)
// [73728  , 147456)  q[144][128], k[144][128]    / w_chunk[144][128] (stage3)
// [147456 , 294912)  vT[512][144] f16
// [294912 , 327680)  wbuf[2][32][256] f16 (double-buffered async W_uv tiles)
#define SMEM_BYTES 327680

#define AS1 __attribute__((address_space(1)))
#define AS3 __attribute__((address_space(3)))

#define WMMA_F16(acc, a, b) \
  (acc) = __builtin_amdgcn_wmma_f32_16x16x32_f16(false, (a), false, (b), (short)0, (acc), false, false)

// A fragment: 16x32 f16 tile from row-major LDS (pitch in halves).
__device__ inline v16h ld_a(const _Float16* A, int pitch, int m0, int k0, int lane) {
  const _Float16* p = A + (m0 + (lane & 15)) * pitch + k0 + ((lane & 16) ? 16 : 0);
  v8h lo = *(const v8h*)p;
  v8h hi = *(const v8h*)(p + 8);
  v16h r;
#pragma unroll
  for (int j = 0; j < 8; ++j) { r[j] = lo[j]; r[8 + j] = hi[j]; }
  return r;
}

// B fragment from [n][k]-major f16 memory (LDS or global), pitch in halves.
__device__ inline v16h ld_bt(const _Float16* BT, int pitch, int n0, int k0, int lane) {
  const _Float16* p = BT + (size_t)(n0 + (lane & 15)) * pitch + k0 + ((lane & 16) ? 16 : 0);
  v8h lo = *(const v8h*)p;
  v8h hi = *(const v8h*)(p + 8);
  v16h r;
#pragma unroll
  for (int j = 0; j < 8; ++j) { r[j] = lo[j]; r[8 + j] = hi[j]; }
  return r;
}

// Guarded B fragment (zero-fill k >= kmax) for the ragged last K-step of the attn GEMM.
__device__ inline v16h ld_bt_guard(const _Float16* BT, int pitch, int n0, int k0, int kmax, int lane) {
  const _Float16* p = BT + (n0 + (lane & 15)) * pitch;
  int kb = k0 + ((lane & 16) ? 16 : 0);
  v16h r;
#pragma unroll
  for (int j = 0; j < 16; ++j) {
    int k = kb + j;
    r[j] = (k < kmax) ? p[k] : (_Float16)0.f;
  }
  return r;
}

// Issue async DMA of one 16KB W_uv tile (32 rows x 256 halves) into LDS.
// Each of 256 threads moves 64 bytes = 4 x b128; INST_OFFSET applies to both addresses.
__device__ inline void async_tile_copy(const _Float16* __restrict__ gsrc,
                                       _Float16* __restrict__ ldst, int tid) {
  AS1 vi4* g = (AS1 vi4*)((const char*)gsrc + tid * 64);
  AS3 vi4* l = (AS3 vi4*)((char*)ldst + tid * 64);
  __builtin_amdgcn_global_load_async_to_lds_b128(g, l, 0,  0);
  __builtin_amdgcn_global_load_async_to_lds_b128(g, l, 16, 0);
  __builtin_amdgcn_global_load_async_to_lds_b128(g, l, 32, 0);
  __builtin_amdgcn_global_load_async_to_lds_b128(g, l, 48, 0);
}

__device__ inline void wait_async_all() {
  asm volatile("s_wait_asynccnt 0x0" ::: "memory");
}

// ---- Prelude: one-time fp32 -> fp16 conversion of W_uv and W_o into workspace ----
__global__ __launch_bounds__(256)
void convert_weights_kernel(const float* __restrict__ W_uv, const float* __restrict__ W_o,
                            _Float16* __restrict__ W_uv_h, _Float16* __restrict__ W_o_h) {
  int i = blockIdx.x * 256 + threadIdx.x;
  if (i < (2 * Em + Sm) * Dm) W_uv_h[i] = (_Float16)W_uv[i];
  if (i < Dm * Em)            W_o_h[i]  = (_Float16)W_o[i];
}

__global__ __launch_bounds__(256)
void rtmcc_fused_kernel(const float* __restrict__ x, const float* __restrict__ gamma,
                        const float* __restrict__ beta, const float* __restrict__ g,
                        const float* __restrict__ res_scale,
                        const _Float16* __restrict__ W_uv_h,
                        const _Float16* __restrict__ W_o_h,
                        _Float16* __restrict__ u_ws, float* __restrict__ out) {
  __shared__ __align__(16) unsigned char smem[SMEM_BYTES];
  _Float16* xn    = (_Float16*)smem;                 // stage 1
  _Float16* kern  = (_Float16*)smem;                 // stage 2+ (overlays xn)
  _Float16* qL    = (_Float16*)(smem + 73728);
  _Float16* kL    = (_Float16*)(smem + 110592);
  _Float16* wch   = (_Float16*)(smem + 73728);       // stage 3 (overlays q/k)
  _Float16* vT    = (_Float16*)(smem + 147456);
  _Float16* wbuf0 = (_Float16*)(smem + 294912);
  _Float16* wbuf1 = (_Float16*)(smem + 311296);

  const int b    = blockIdx.x;
  const int tid  = threadIdx.x;
  const int wv   = tid >> 5;
  const int lane = tid & 31;

  // ---------- Stage 0: RMS-norm x -> xn (f16, LDS); prefill W tile 0 ----------
  async_tile_copy(W_uv_h, wbuf0, tid);               // chunk 0 DMA overlaps stage 0
  {
    const float gval = g[0];
    for (int t = wv; t < Tm; t += 8) {
      const float* xr = x + ((size_t)b * Tm + t) * Dm + lane * 8;
      float4 f0 = *(const float4*)xr;
      float4 f1 = *(const float4*)(xr + 4);
      float ss = f0.x*f0.x + f0.y*f0.y + f0.z*f0.z + f0.w*f0.w
               + f1.x*f1.x + f1.y*f1.y + f1.z*f1.z + f1.w*f1.w;
#pragma unroll
      for (int off = 16; off > 0; off >>= 1) ss += __shfl_xor(ss, off, 32);
      float norm = sqrtf(ss) * 0.0625f;              // ||x|| * D^-0.5
      float sc = gval / fmaxf(norm, 1e-5f);
      v8h h;
      h[0]=(_Float16)(f0.x*sc); h[1]=(_Float16)(f0.y*sc);
      h[2]=(_Float16)(f0.z*sc); h[3]=(_Float16)(f0.w*sc);
      h[4]=(_Float16)(f1.x*sc); h[5]=(_Float16)(f1.y*sc);
      h[6]=(_Float16)(f1.z*sc); h[7]=(_Float16)(f1.w*sc);
      *(v8h*)(xn + t * Dm + lane * 8) = h;
    }
    for (int i = tid; i < (TP - Tm) * Dm / 8; i += 256) {   // zero pad rows
      v8h z;
#pragma unroll
      for (int j = 0; j < 8; ++j) z[j] = (_Float16)0.f;
      ((v8h*)(xn + Tm * Dm))[i] = z;
    }
  }
  wait_async_all();
  __syncthreads();

  // ---------- Stage 1: uv = silu(xn @ W_uv^T) -> u(global), vT(LDS), q/k(LDS) ----------
  for (int ch = 0; ch < 36; ++ch) {                  // 36 chunks of 32 output cols
    if (ch + 1 < 36)                                 // DMA next tile while computing
      async_tile_copy(W_uv_h + (size_t)(ch + 1) * 32 * Dm,
                      ((ch + 1) & 1) ? wbuf1 : wbuf0, tid);
    const _Float16* wb = (ch & 1) ? wbuf1 : wbuf0;

    for (int tix = wv; tix < 18; tix += 8) {         // 2 N-tiles x 9 M-tiles
      int ntl = tix / 9, mtl = tix % 9;
      v8f acc = {};
#pragma unroll
      for (int ks = 0; ks < 8; ++ks) {
        v16h a  = ld_a (xn, Dm, mtl * 16, ks * 32, lane);
        v16h bf = ld_bt(wb, Dm, ntl * 16, ks * 32, lane);
        WMMA_F16(acc, a, bf);
      }
      int e0 = ch * 32 + ntl * 16;
      int n  = e0 + (lane & 15);
      int mb = mtl * 16 + ((lane & 16) ? 8 : 0);
      float ga0 = 0.f, be0 = 0.f, ga1 = 0.f, be1 = 0.f;
      if (e0 >= 2 * Em) {
        int s = n - 2 * Em;
        ga0 = gamma[s];      be0 = beta[s];
        ga1 = gamma[Sm + s]; be1 = beta[Sm + s];
      }
#pragma unroll
      for (int r = 0; r < 8; ++r) {
        int t = mb + r;
        float v = acc[r];
        v = v / (1.f + __expf(-v));                  // silu
        if (e0 < Em) {                               // u -> global workspace
          if (t < Tm) u_ws[((size_t)b * Tm + t) * Em + n] = (_Float16)v;
        } else if (e0 < 2 * Em) {                    // v -> transposed LDS
          vT[(n - Em) * TP + t] = (_Float16)v;
        } else {                                     // base -> q, k
          int s = n - 2 * Em;
          qL[t * Sm + s] = (_Float16)(v * ga0 + be0);
          kL[t * Sm + s] = (_Float16)(v * ga1 + be1);
        }
      }
    }
    wait_async_all();
    __syncthreads();
  }

  // ---------- Stage 2: kernel = relu(q k^T / sqrt(S))^2 -> kern[144][160] ----------
  for (int i = tid; i < TP * (KERNP - TP); i += 256) {   // zero padded cols [144,160)
    int row = i / (KERNP - TP), c = TP + i % (KERNP - TP);
    kern[row * KERNP + c] = (_Float16)0.f;
  }
  for (int tix = wv; tix < 81; tix += 8) {           // 9x9 tiles
    int mtl = tix / 9, ntl = tix % 9;
    v8f acc = {};
#pragma unroll
    for (int ks = 0; ks < 4; ++ks) {
      v16h a  = ld_a (qL, Sm, mtl * 16, ks * 32, lane);
      v16h bf = ld_bt(kL, Sm, ntl * 16, ks * 32, lane);
      WMMA_F16(acc, a, bf);
    }
    int n  = ntl * 16 + (lane & 15);
    int mb = mtl * 16 + ((lane & 16) ? 8 : 0);
#pragma unroll
    for (int r = 0; r < 8; ++r) {
      float s = fmaxf(acc[r] * RSQRT_S, 0.f);
      kern[(mb + r) * KERNP + n] = (_Float16)(s * s);
    }
  }
  __syncthreads();

  // ---------- Stage 3: attn = kernel @ v ; w = u*attn ; out += w @ W_o^T ----------
  v8f oacc[18];
#pragma unroll
  for (int i = 0; i < 18; ++i) oacc[i] = (v8f){};

  for (int ch = 0; ch < 4; ++ch) {                   // 4 chunks of 128 E-columns
    for (int tix = wv; tix < 72; tix += 8) {         // 9 M-tiles x 8 local N-tiles
      int mtl = tix / 8, ntl = tix % 8;
      int e0 = ch * 128 + ntl * 16;
      v8f acc = {};
#pragma unroll
      for (int ks = 0; ks < 5; ++ks) {               // K = T padded to 160
        v16h a  = ld_a(kern, KERNP, mtl * 16, ks * 32, lane);
        v16h bf = (ks < 4) ? ld_bt(vT, TP, e0, ks * 32, lane)
                           : ld_bt_guard(vT, TP, e0, 128, TP, lane);
        WMMA_F16(acc, a, bf);
      }
      int n  = lane & 15;
      int mb = mtl * 16 + ((lane & 16) ? 8 : 0);
      int eg = e0 + n;
#pragma unroll
      for (int r = 0; r < 8; ++r) {
        int t  = mb + r;
        int tc = (t < Tm) ? t : (Tm - 1);
        float uval = (float)u_ws[((size_t)b * Tm + tc) * Em + eg];
        if (t >= Tm) uval = 0.f;
        wch[t * 128 + ntl * 16 + n] = (_Float16)(uval * acc[r]);
      }
    }
    __syncthreads();
    // partial out-GEMM: oacc += w_chunk @ W_o[:, chunk]^T  (B direct from f16 global)
    for (int i = 0; i < 18; ++i) {
      int tix = wv + i * 8;                          // 0..143
      int mtl = tix / 16, ntl = tix % 16;
      v8f acc = oacc[i];
#pragma unroll
      for (int ks = 0; ks < 4; ++ks) {
        v16h a  = ld_a (wch,   128, mtl * 16, ks * 32, lane);
        v16h bf = ld_bt(W_o_h, Em,  ntl * 16, ch * 128 + ks * 32, lane);
        WMMA_F16(acc, a, bf);
      }
      oacc[i] = acc;
    }
    __syncthreads();                                 // wch reused next chunk
  }

  // ---------- Epilogue: out = x * res_scale + oacc ----------
  for (int i = 0; i < 18; ++i) {
    int tix = wv + i * 8;
    int mtl = tix / 16, ntl = tix % 16;
    int n  = ntl * 16 + (lane & 15);
    int mb = mtl * 16 + ((lane & 16) ? 8 : 0);
    float rs = res_scale[n];
#pragma unroll
    for (int r = 0; r < 8; ++r) {
      int t = mb + r;
      if (t < Tm) {
        size_t idx = ((size_t)b * Tm + t) * Dm + n;
        out[idx] = x[idx] * rs + oacc[i][r];
      }
    }
  }
}

extern "C" void kernel_launch(void* const* d_in, const int* in_sizes, int n_in,
                              void* d_out, int out_size, void* d_ws, size_t ws_size,
                              hipStream_t stream) {
  const float* x         = (const float*)d_in[0];
  const float* W_uv      = (const float*)d_in[1];
  const float* W_o       = (const float*)d_in[2];
  const float* gamma     = (const float*)d_in[3];
  const float* beta      = (const float*)d_in[4];
  const float* g         = (const float*)d_in[5];
  const float* res_scale = (const float*)d_in[6];
  float* out = (float*)d_out;
  (void)in_sizes; (void)n_in; (void)out_size; (void)ws_size;

  // workspace layout: u (B*T*E f16), then f16 weight copies (256B-aligned)
  const size_t u_bytes = (size_t)1024 * Tm * Em * sizeof(_Float16);  // 139,460,608
  _Float16* u_ws   = (_Float16*)d_ws;
  _Float16* W_uv_h = (_Float16*)((char*)d_ws + u_bytes);             // 589,824 B
  _Float16* W_o_h  = (_Float16*)((char*)d_ws + u_bytes + 589824);    // 262,144 B

  convert_weights_kernel<<<1152, 256, 0, stream>>>(W_uv, W_o, W_uv_h, W_o_h);
  rtmcc_fused_kernel<<<1024, 256, 0, stream>>>(x, gamma, beta, g, res_scale,
                                               W_uv_h, W_o_h, u_ws, out);
}
